// MultiheadAttention_78323023610561
// MI455X (gfx1250) — compile-verified, matrix operand order checked by reading, not run on hardware
//
#include <hip/hip_runtime.h>

// MultiheadAttention forward for MI455X (gfx1250, wave32, WMMA + TDM).
// B=2, T=2048, E=1024, H=16, D=64.  All GEMMs in bf16 WMMA (f32 accum).
//
// Pipeline:
//   gemm<0>: Q = query @ Wq^T  -> bf16 [B,H,T,D]
//   gemm<0>: K = key_  @ Wk^T  -> bf16 [B,H,T,D]
//   gemm<2>: Vt= value @ Wv^T  -> bf16 [B,H,D,T]   (transposed for PV B-frags)
//   attn   : flash attention; K/V tiles double-buffered via Tensor Data Mover
//   gemm<3>: out = attn @ Wo^T -> f32  [B,T,E]

#define TT 2048
#define EE 1024
#define HH 16
#define DD 64
#define SCALE 0.125f

typedef __attribute__((ext_vector_type(16))) __bf16   v16bf;
typedef __attribute__((ext_vector_type(8)))  float    v8f;
typedef __attribute__((ext_vector_type(4)))  unsigned v4u;
typedef __attribute__((ext_vector_type(8)))  unsigned v8u;

union FragU { uint4 q[2]; v16bf v; };

__device__ __forceinline__ unsigned pack2bf(float a, float b) {
    unsigned ua = __float_as_uint(a), ub = __float_as_uint(b);
    ua += 0x7FFFu + ((ua >> 16) & 1u);          // round-to-nearest-even
    ub += 0x7FFFu + ((ub >> 16) & 1u);
    return (ua >> 16) | (ub & 0xFFFF0000u);
}
__device__ __forceinline__ unsigned short f2bf(float a) {
    unsigned u = __float_as_uint(a);
    u += 0x7FFFu + ((u >> 16) & 1u);
    return (unsigned short)(u >> 16);
}
// reduce across the 16 lanes that hold one C-matrix row (xor 1,2,4,8 stays in group)
__device__ __forceinline__ float red_max16(float v) {
    v = fmaxf(v, __int_as_float(__builtin_amdgcn_ds_swizzle(__float_as_int(v), 0x041f)));
    v = fmaxf(v, __int_as_float(__builtin_amdgcn_ds_swizzle(__float_as_int(v), 0x081f)));
    v = fmaxf(v, __int_as_float(__builtin_amdgcn_ds_swizzle(__float_as_int(v), 0x101f)));
    v = fmaxf(v, __int_as_float(__builtin_amdgcn_ds_swizzle(__float_as_int(v), 0x201f)));
    return v;
}
__device__ __forceinline__ float red_sum16(float v) {
    v += __int_as_float(__builtin_amdgcn_ds_swizzle(__float_as_int(v), 0x041f));
    v += __int_as_float(__builtin_amdgcn_ds_swizzle(__float_as_int(v), 0x081f));
    v += __int_as_float(__builtin_amdgcn_ds_swizzle(__float_as_int(v), 0x101f));
    v += __int_as_float(__builtin_amdgcn_ds_swizzle(__float_as_int(v), 0x201f));
    return v;
}

// ---------------------------------------------------------------------------
// TDM: issue a 2D tensor_load_to_lds for a bf16 tile (D# per ISA ch. 8).
// All descriptor fields are block-uniform -> "s" constraints (SGPR groups).
// ---------------------------------------------------------------------------
__device__ __forceinline__ void tdm_load_2d_bf16(unsigned lds_off, const void* gptr,
                                                 unsigned tensor_d0, unsigned tensor_d1,
                                                 unsigned tile_d0, unsigned tile_d1,
                                                 unsigned stride0) {
    unsigned long long ga = (unsigned long long)(uintptr_t)gptr;
    v4u g0;
    g0[0] = 1u;                                              // count=1, user D#
    g0[1] = lds_off;                                         // lds_addr (bytes)
    g0[2] = (unsigned)ga;                                    // global_addr[31:0]
    g0[3] = (unsigned)((ga >> 32) & 0x01FFFFFFu) | (2u << 30); // addr[56:32]|type=2
    v8u g1;
    g1[0] = (1u << 16);                                      // data_size=1 (2 bytes)
    g1[1] = (tensor_d0 & 0xFFFFu) << 16;                     // tensor_dim0[15:0]
    g1[2] = (tensor_d0 >> 16) | ((tensor_d1 & 0xFFFFu) << 16);
    g1[3] = (tensor_d1 >> 16) | (tile_d0 << 16);             // tile_dim0
    g1[4] = tile_d1;                                         // tile_dim1, tile_dim2=0
    g1[5] = stride0;                                         // dim0_stride[31:0]
    g1[6] = 0u;
    g1[7] = 0u;
    asm volatile("tensor_load_to_lds %0, %1" :: "s"(g0), "s"(g1) : "memory");
}

// ---------------------------------------------------------------------------
// GEMM: C[4096, 1024] = A[4096,1024] * W[1024,1024]^T
// MODE 0: A=f32, out bf16 -> [B,H,T,D]      (Q and K projections)
// MODE 2: A=f32, out bf16 -> [B,H,D,T]      (V projection, transposed)
// MODE 3: A=bf16, out f32 -> [4096,1024]    (output projection)
// Block tile M=64,N=128, 4 waves (each 64x32).  Global loads for chunk k+1
// are register-prefetched during the WMMA phase of chunk k (latency hiding).
// ---------------------------------------------------------------------------
template <int MODE>
__global__ __launch_bounds__(128)
void gemm_k(const void* __restrict__ Ag, const float* __restrict__ W,
            void* __restrict__ Out) {
    __shared__ uint4 sA[4 * 32 * 2];   // A frags: [msub][lane][2x16B]
    __shared__ uint4 sB[8 * 32 * 2];   // B frags: [nsub][lane][2x16B]

    const int tid  = threadIdx.x;
    const int wave = tid >> 5, lane = tid & 31;
    const int half = lane >> 4, ln = lane & 15;
    const int m0 = blockIdx.y * 64;
    const int n0 = blockIdx.x * 128;

    v8f acc[4][2];
#pragma unroll
    for (int i = 0; i < 4; ++i)
#pragma unroll
        for (int j = 0; j < 2; ++j)
#pragma unroll
            for (int r = 0; r < 8; ++r) acc[i][j][r] = 0.f;

    // register prefetch buffers (one A fragment-lane + two B fragment-lanes)
    float4 fa[4];
    uint4  ua[2];
    float4 fb[2][4];

    auto prefetch = [&](int kk) {
        const int l = tid & 31;
        const int kb = kk + ((l >> 4) << 3);
        {
            const int msub = tid >> 5;
            const int m = m0 + msub * 16 + (l & 15);
            if (MODE == 3) {
                const unsigned short* ar = (const unsigned short*)Ag + (size_t)m * EE;
                ua[0] = *reinterpret_cast<const uint4*>(ar + kb);
                ua[1] = *reinterpret_cast<const uint4*>(ar + kb + 16);
            } else {
                const float* ar = (const float*)Ag + (size_t)m * EE;
                fa[0] = *reinterpret_cast<const float4*>(ar + kb);
                fa[1] = *reinterpret_cast<const float4*>(ar + kb + 4);
                fa[2] = *reinterpret_cast<const float4*>(ar + kb + 16);
                fa[3] = *reinterpret_cast<const float4*>(ar + kb + 20);
            }
        }
#pragma unroll
        for (int i = 0; i < 2; ++i) {
            const int id = tid + i * 128;
            const int nsub = id >> 5, ll = id & 31;
            const int n = n0 + nsub * 16 + (ll & 15);
            const int kbb = kk + ((ll >> 4) << 3);
            const float* wr = W + (size_t)n * EE;
            fb[i][0] = *reinterpret_cast<const float4*>(wr + kbb);
            fb[i][1] = *reinterpret_cast<const float4*>(wr + kbb + 4);
            fb[i][2] = *reinterpret_cast<const float4*>(wr + kbb + 16);
            fb[i][3] = *reinterpret_cast<const float4*>(wr + kbb + 20);
        }
    };

    prefetch(0);

    for (int kk = 0; kk < EE; kk += 32) {
        __syncthreads();                      // previous compute done with LDS
        {   // store prefetched chunk to LDS (convert f32->bf16 if needed)
            const int msub = tid >> 5, l = tid & 31;
            uint4 r0, r1;
            if (MODE == 3) {
                r0 = ua[0]; r1 = ua[1];
            } else {
                r0 = make_uint4(pack2bf(fa[0].x, fa[0].y), pack2bf(fa[0].z, fa[0].w),
                                pack2bf(fa[1].x, fa[1].y), pack2bf(fa[1].z, fa[1].w));
                r1 = make_uint4(pack2bf(fa[2].x, fa[2].y), pack2bf(fa[2].z, fa[2].w),
                                pack2bf(fa[3].x, fa[3].y), pack2bf(fa[3].z, fa[3].w));
            }
            sA[(msub * 32 + l) * 2 + 0] = r0;
            sA[(msub * 32 + l) * 2 + 1] = r1;
#pragma unroll
            for (int i = 0; i < 2; ++i) {
                const int id = tid + i * 128;
                const int nsub = id >> 5, ll = id & 31;
                sB[(nsub * 32 + ll) * 2 + 0] =
                    make_uint4(pack2bf(fb[i][0].x, fb[i][0].y), pack2bf(fb[i][0].z, fb[i][0].w),
                               pack2bf(fb[i][1].x, fb[i][1].y), pack2bf(fb[i][1].z, fb[i][1].w));
                sB[(nsub * 32 + ll) * 2 + 1] =
                    make_uint4(pack2bf(fb[i][2].x, fb[i][2].y), pack2bf(fb[i][2].z, fb[i][2].w),
                               pack2bf(fb[i][3].x, fb[i][3].y), pack2bf(fb[i][3].z, fb[i][3].w));
            }
        }
        __syncthreads();
        if (kk + 32 < EE) prefetch(kk + 32);  // vmem issues overlap the WMMAs below
#pragma unroll
        for (int msub = 0; msub < 4; ++msub) {
            FragU a;
            a.q[0] = sA[(msub * 32 + lane) * 2 + 0];
            a.q[1] = sA[(msub * 32 + lane) * 2 + 1];
#pragma unroll
            for (int j = 0; j < 2; ++j) {
                const int nsub = wave * 2 + j;
                FragU b;
                b.q[0] = sB[(nsub * 32 + lane) * 2 + 0];
                b.q[1] = sB[(nsub * 32 + lane) * 2 + 1];
                acc[msub][j] = __builtin_amdgcn_wmma_f32_16x16x32_bf16(
                    false, a.v, false, b.v, (short)0, acc[msub][j], false, false);
            }
        }
    }
    // epilogue (C layout: lane -> n = ln, vgpr r -> m = half*8+r)
#pragma unroll
    for (int msub = 0; msub < 4; ++msub)
#pragma unroll
        for (int j = 0; j < 2; ++j)
#pragma unroll
            for (int r = 0; r < 8; ++r) {
                const int m = m0 + msub * 16 + half * 8 + r;
                const int n = n0 + (wave * 2 + j) * 16 + ln;
                const float v = acc[msub][j][r];
                if (MODE == 3) {
                    ((float*)Out)[(size_t)m * EE + n] = v;
                } else {
                    const int b = m >> 11, t = m & (TT - 1);
                    const int h = n >> 6, d = n & (DD - 1);
                    const unsigned short bv = f2bf(v);
                    if (MODE == 2)   // [B,H,D,T]
                        ((unsigned short*)Out)[(((size_t)(b * HH + h) * DD + d) * TT) + t] = bv;
                    else             // [B,H,T,D]
                        ((unsigned short*)Out)[(((size_t)(b * HH + h) * TT + t) * DD) + d] = bv;
                }
            }
}

// ---------------------------------------------------------------------------
// Flash attention: one block = (b,h) x 64 query rows; 4 waves x 16 rows.
// K/Vt tiles are DMA'd by the TDM into double-buffered LDS: while iteration i
// computes on buffer p, the DMA for iteration i+1 streams into buffer p^1
// (TENSORcnt retires in order, so s_wait_tensorcnt 2 == "tile i landed").
// ---------------------------------------------------------------------------
__global__ __launch_bounds__(128)
void attn_k(const unsigned short* __restrict__ Q,
            const unsigned short* __restrict__ K,
            const unsigned short* __restrict__ Vt,
            const float* __restrict__ mask,
            unsigned short* __restrict__ AttnC) {
    __shared__ __align__(16) unsigned short sKt[2][64 * 64];      // 2 x 8 KB
    __shared__ __align__(16) unsigned short sVt[2][64 * 64];      // 2 x 8 KB
    __shared__ __align__(16) unsigned short sP[4 * 2 * 32 * 16];  // 8 KB P frags

    const int tid  = threadIdx.x;
    const int wave = tid >> 5, lane = tid & 31;
    const int half = lane >> 4, ln = lane & 15;
    const int bh = blockIdx.x;                  // b*16 + h
    const int t0 = blockIdx.y * 64 + wave * 16; // this wave's 16 query rows

    const unsigned short* Qb = Q  + (size_t)bh * TT * DD;
    const unsigned short* Kb = K  + (size_t)bh * TT * DD;
    const unsigned short* Vb = Vt + (size_t)bh * DD * TT;

    // generic->LDS byte offset (low 32 bits of the flat LDS address)
    unsigned sK_off[2] = { (unsigned)(uintptr_t)&sKt[0][0], (unsigned)(uintptr_t)&sKt[1][0] };
    unsigned sV_off[2] = { (unsigned)(uintptr_t)&sVt[0][0], (unsigned)(uintptr_t)&sVt[1][0] };

    // Q fragments (A layout), kept in registers: lane m = ln, k split by half
    FragU qa[2];
    {
        const unsigned short* qr = Qb + (size_t)(t0 + ln) * DD;
        const int kb = half << 3;
#pragma unroll
        for (int kc = 0; kc < 2; ++kc) {
            qa[kc].q[0] = *reinterpret_cast<const uint4*>(qr + kc * 32 + kb);
            qa[kc].q[1] = *reinterpret_cast<const uint4*>(qr + kc * 32 + kb + 16);
        }
    }

    v8f O[4];
    float rm[8], rs[8];
#pragma unroll
    for (int r = 0; r < 8; ++r) { rm[r] = -3.0e38f; rs[r] = 0.f; }
#pragma unroll
    for (int d = 0; d < 4; ++d)
#pragma unroll
        for (int r = 0; r < 8; ++r) O[d][r] = 0.f;

    // prologue: DMA first tile pair into buffer 0
    if (wave == 0) {
        tdm_load_2d_bf16(sK_off[0], Kb, DD, TT, DD, 64, DD);
        tdm_load_2d_bf16(sV_off[0], Vb, TT, DD, 64, DD, TT);
    }

    const int NIT = TT / 64;
    for (int it = 0; it < NIT; ++it) {
        const int p = it & 1;
        const int s0 = it * 64;
        if (wave == 0) {
            if (it + 1 < NIT) {
                const int s1 = s0 + 64;   // prefetch next tile pair into buffer p^1
                tdm_load_2d_bf16(sK_off[p ^ 1], Kb + (size_t)s1 * DD, DD, TT, DD, 64, DD);
                tdm_load_2d_bf16(sV_off[p ^ 1], Vb + (size_t)s1, TT, DD, 64, DD, TT);
                __builtin_amdgcn_s_wait_tensorcnt(2);   // tile `it` complete
            } else {
                __builtin_amdgcn_s_wait_tensorcnt(0);
            }
        }
        __syncthreads();                 // tile `it` visible to all waves

        const uint4* ktile = reinterpret_cast<const uint4*>(sKt[p]);
        const uint4* vtile = reinterpret_cast<const uint4*>(sVt[p]);

        // ---- scores: 16x64 tile = 4 nsub x (2 WMMA over D) ----
        v8f S[4];
#pragma unroll
        for (int nsub = 0; nsub < 4; ++nsub) {
#pragma unroll
            for (int r = 0; r < 8; ++r) S[nsub][r] = 0.f;
#pragma unroll
            for (int kc = 0; kc < 2; ++kc) {
                FragU b;   // B-frag: row s = nsub*16+ln, 2x16B at d = kc*32+half*8
                const int base = (nsub * 16 + ln) * 8 + kc * 4 + half;
                b.q[0] = ktile[base];
                b.q[1] = ktile[base + 2];
                S[nsub] = __builtin_amdgcn_wmma_f32_16x16x32_bf16(
                    false, qa[kc].v, false, b.v, (short)0, S[nsub], false, false);
            }
        }
        // ---- scale + additive mask ----
#pragma unroll
        for (int nsub = 0; nsub < 4; ++nsub) {
            const int s = s0 + nsub * 16 + ln;
#pragma unroll
            for (int r = 0; r < 8; ++r)
                S[nsub][r] = S[nsub][r] * SCALE + mask[(size_t)(t0 + half * 8 + r) * TT + s];
        }
        // ---- online softmax ----
        float bm[8];
#pragma unroll
        for (int r = 0; r < 8; ++r) {
            float v = fmaxf(fmaxf(S[0][r], S[1][r]), fmaxf(S[2][r], S[3][r]));
            bm[r] = red_max16(v);
        }
#pragma unroll
        for (int r = 0; r < 8; ++r) {
            const float mn = fmaxf(rm[r], bm[r]);
            const float c = __expf(rm[r] - mn);
            rm[r] = mn;
            rs[r] *= c;
#pragma unroll
            for (int d = 0; d < 4; ++d) O[d][r] *= c;
        }
        // P = exp(S - m): write to LDS in A-frag layout (C->A transpose bounce)
#pragma unroll
        for (int nsub = 0; nsub < 4; ++nsub) {
            const int sl = nsub * 16 + ln;      // local s column 0..63
            const int kc = sl >> 5;
            const int w = sl & 31;
            const int hh = (w >> 3) & 1;
            const int e = (w & 7) + ((w >> 4) << 3);
#pragma unroll
            for (int r = 0; r < 8; ++r) {
                const float p2 = __expf(S[nsub][r] - rm[r]);
                S[nsub][r] = p2;
                const int lp = (half * 8 + r) + (hh << 4);
                sP[((wave * 2 + kc) * 32 + lp) * 16 + e] = f2bf(p2);
            }
        }
#pragma unroll
        for (int r = 0; r < 8; ++r)
            rs[r] += red_sum16(S[0][r] + S[1][r] + S[2][r] + S[3][r]);

        // ---- PV: O += P (16x64) x V (64x64) ----
#pragma unroll
        for (int kc = 0; kc < 2; ++kc) {
            FragU pa;
            pa.q[0] = reinterpret_cast<const uint4*>(sP)[((wave * 2 + kc) * 32 + lane) * 2 + 0];
            pa.q[1] = reinterpret_cast<const uint4*>(sP)[((wave * 2 + kc) * 32 + lane) * 2 + 1];
#pragma unroll
            for (int dsub = 0; dsub < 4; ++dsub) {
                FragU b;   // B-frag: row d = dsub*16+ln, 2x16B at s = kc*32+half*8
                const int base = (dsub * 16 + ln) * 8 + kc * 4 + half;
                b.q[0] = vtile[base];
                b.q[1] = vtile[base + 2];
                O[dsub] = __builtin_amdgcn_wmma_f32_16x16x32_bf16(
                    false, pa.v, false, b.v, (short)0, O[dsub], false, false);
            }
        }
        __syncthreads();                 // all done with buffer p before it is
                                         // overwritten by the DMA issued at it+1
    }
    // ---- normalize and store attn context [B,T,H*D] bf16 ----
    const int b = bh >> 4, h = bh & 15;
#pragma unroll
    for (int r = 0; r < 8; ++r) {
        const float inv = 1.0f / rs[r];
        const int t = t0 + half * 8 + r;
#pragma unroll
        for (int dsub = 0; dsub < 4; ++dsub) {
            const int d = dsub * 16 + ln;
            AttnC[(size_t)(b * TT + t) * EE + h * DD + d] = f2bf(O[dsub][r] * inv);
        }
    }
}

extern "C" void kernel_launch(void* const* d_in, const int* in_sizes, int n_in,
                              void* d_out, int out_size, void* d_ws, size_t ws_size,
                              hipStream_t stream) {
    (void)in_sizes; (void)n_in; (void)out_size; (void)ws_size;
    const float* query = (const float*)d_in[0];
    const float* key_  = (const float*)d_in[1];
    const float* value = (const float*)d_in[2];
    const float* amask = (const float*)d_in[3];
    const float* Wq    = (const float*)d_in[4];
    const float* Wk    = (const float*)d_in[5];
    const float* Wv    = (const float*)d_in[6];
    const float* Wo    = (const float*)d_in[7];

    const size_t NQKV = (size_t)2 * HH * TT * DD;       // 4,194,304 elems
    unsigned short* Qw  = (unsigned short*)d_ws;
    unsigned short* Kw  = Qw + NQKV;
    unsigned short* Vtw = Kw + NQKV;
    unsigned short* Cw  = Vtw + NQKV;                   // total 32 MiB bf16

    const dim3 gg(EE / 128, (2 * TT) / 64);             // (8, 64)
    gemm_k<0><<<gg, 128, 0, stream>>>(query, Wq, Qw);
    gemm_k<0><<<gg, 128, 0, stream>>>(key_,  Wk, Kw);
    gemm_k<2><<<gg, 128, 0, stream>>>(value, Wv, Vtw);
    attn_k<<<dim3(2 * HH, TT / 64), 128, 0, stream>>>(Qw, Kw, Vtw, amask, Cw);
    gemm_k<3><<<gg, 128, 0, stream>>>(Cw, Wo, (float*)d_out);
}
